// RLIPv2_BiAttentionBlockForCheckpoint_36558761623764
// MI455X (gfx1250) — compile-verified, matrix operand order checked by reading, not run on hardware
//
#include <hip/hip_runtime.h>

typedef __attribute__((ext_vector_type(16))) _Float16 v16h;
typedef __attribute__((ext_vector_type(8)))  float    v8f;

#define B_      8
#define TGT     8192
#define SRC     64
#define V_DIM   256
#define L_DIM   768
#define EMBED   1024
#define HEADS   16
#define HEAD_DIM 64
#define QSCALE  0.125f   // HEAD_DIM^-0.5
#define LDP     72       // padded LDS row stride (halfs) to spread banks
#define TSPLIT  4        // T-range split for l-side attention

// ---------------------------------------------------------------------------
// WMMA helper (CDNA5 16x16x32 f16 -> f32)
// ---------------------------------------------------------------------------
__device__ __forceinline__ v8f wmma16(v16h a, v16h b, v8f c) {
  return __builtin_amdgcn_wmma_f32_16x16x32_f16(false, a, false, b, (short)0, c,
                                                false, false);
}

// Load a 16(m) x 32(k) f16 operand tile with the documented A/B lane layout:
//   lanes 0-15: m=lane,   K chunks {0..7, 16..23}
//   lanes 16-31: m=lane-16, K chunks {8..15, 24..31}
// element (m,k) at p[m*ldm + k*ldk].  Works for global and LDS pointers.
__device__ __forceinline__ v16h load_tile(const _Float16* p, int ldm, int ldk) {
  const int lane = threadIdx.x & 31;
  const int half = lane >> 4;
  const int m    = lane & 15;
  const _Float16* pm = p + m * ldm + half * 8 * ldk;
  v16h r;
#pragma unroll
  for (int i = 0; i < 8; ++i) {
    const int kb = ((i & 4) ? 16 : 0) + 2 * (i & 3);
    r[2 * i]     = pm[kb * ldk];
    r[2 * i + 1] = pm[(kb + 1) * ldk];
  }
  return r;
}

// ---------------------------------------------------------------------------
// Cross-lane reductions via ds_swizzle (single VDS op per step; group-of-32
// pattern: new_lane = lane ^ XORMASK, imm = (xor<<10) | and_mask 0x1f)
// ---------------------------------------------------------------------------
template <int XORMASK>
__device__ __forceinline__ float lane_xor(float v) {
  constexpr int imm = 0x1f | (XORMASK << 10);
  return __int_as_float(__builtin_amdgcn_ds_swizzle(__float_as_int(v), imm));
}
__device__ __forceinline__ float redmax16(float v) {   // within 16-lane halves
  v = fmaxf(v, lane_xor<1>(v));
  v = fmaxf(v, lane_xor<2>(v));
  v = fmaxf(v, lane_xor<4>(v));
  v = fmaxf(v, lane_xor<8>(v));
  return v;
}
__device__ __forceinline__ float redsum16(float v) {
  v += lane_xor<1>(v);
  v += lane_xor<2>(v);
  v += lane_xor<4>(v);
  v += lane_xor<8>(v);
  return v;
}
__device__ __forceinline__ float redsum32(float v) {   // full wave
  v = redsum16(v);
  v += lane_xor<16>(v);
  return v;
}

// ---------------------------------------------------------------------------
// Weight conversion: W[K,N] f32 -> Wt[N,K] f16 (transposed so WMMA B-operand
// loads are contiguous K-pairs)
// ---------------------------------------------------------------------------
__global__ void convert_wT_kernel(const float* __restrict__ in,
                                  _Float16* __restrict__ out, int K, int N) {
  const int idx = blockIdx.x * blockDim.x + threadIdx.x;
  if (idx >= K * N) return;
  const int k = idx / N, n = idx - k * N;
  out[(size_t)n * K + k] = (_Float16)in[idx];
}

// ---------------------------------------------------------------------------
// LayerNorm, one wave per row. Optionally emits f16(y) and f16(y + pos).
// ---------------------------------------------------------------------------
template <int DIM>
__global__ void layernorm_kernel(const float* __restrict__ x,
                                 const float* __restrict__ pos,
                                 const float* __restrict__ w,
                                 const float* __restrict__ bb,
                                 _Float16* __restrict__ outn,
                                 _Float16* __restrict__ outq, int rows) {
  const int row = blockIdx.x * (blockDim.x >> 5) + (threadIdx.x >> 5);
  if (row >= rows) return;
  const int lane = threadIdx.x & 31;
  constexpr int PER = DIM / 32;
  float vals[PER];
  const float* xr = x + (size_t)row * DIM;
  float s = 0.f;
#pragma unroll
  for (int i = 0; i < PER; ++i) { vals[i] = xr[lane + i * 32]; s += vals[i]; }
  s = redsum32(s);
  const float mu = s * (1.f / DIM);
  float v2 = 0.f;
#pragma unroll
  for (int i = 0; i < PER; ++i) { const float d = vals[i] - mu; v2 += d * d; }
  v2 = redsum32(v2);
  const float rstd = rsqrtf(v2 * (1.f / DIM) + 1e-5f);
#pragma unroll
  for (int i = 0; i < PER; ++i) {
    const int c = lane + i * 32;
    const float y = (vals[i] - mu) * rstd * w[c] + bb[c];
    outn[(size_t)row * DIM + c] = (_Float16)y;
    if (outq)
      outq[(size_t)row * DIM + c] = (_Float16)(y + pos[(size_t)row * DIM + c]);
  }
}

// ---------------------------------------------------------------------------
// Generic WMMA GEMM: C[M,N] = A[M,K] @ Wt[N,K]^T
// Each wave computes a 16(M) x 64(N) strip: one A tile feeds 4 WMMAs.
// MODE 0: outh[m,n] = f16((acc + bias[n]) * scale)
// MODE 1: outf[m,n] = res[m,n] + gamma[n] * (acc + bias[n])
// 8 waves/block stacked over M. grid = (M/128, N/64)
// ---------------------------------------------------------------------------
template <int MODE>
__global__ __launch_bounds__(256)
void gemm_wmma_kernel(const _Float16* __restrict__ A,
                      const _Float16* __restrict__ Wt,
                      const float* __restrict__ bias, int M, int N, int K,
                      _Float16* __restrict__ outh, float* __restrict__ outf,
                      const float* __restrict__ res,
                      const float* __restrict__ gamma, float scale) {
  const int wave = threadIdx.x >> 5;
  const int m0   = (blockIdx.x * 8 + wave) * 16;
  const int n0   = blockIdx.y * 64;
  const int lane = threadIdx.x & 31;
  const int half = lane >> 4;
  v8f z = {};
  v8f acc[4] = {z, z, z, z};
  for (int k0 = 0; k0 < K; k0 += 32) {
    v16h a = load_tile(A + (size_t)m0 * K + k0, K, 1);
    if (k0 + 32 < K)   // prefetch this lane's next A K-chunk (global_prefetch_b8)
      __builtin_prefetch(A + (size_t)(m0 + (lane & 15)) * K + k0 + 32, 0, 3);
#pragma unroll
    for (int j = 0; j < 4; ++j) {
      v16h b = load_tile(Wt + (size_t)(n0 + j * 16) * K + k0, K, 1);
      acc[j] = wmma16(a, b, acc[j]);
    }
  }
#pragma unroll
  for (int j = 0; j < 4; ++j) {
    const int n = n0 + j * 16 + (lane & 15);
    const float bn = bias[n];
    if (MODE == 0) {
#pragma unroll
      for (int r = 0; r < 8; ++r) {
        const int m = m0 + r + half * 8;
        outh[(size_t)m * N + n] = (_Float16)((acc[j][r] + bn) * scale);
      }
    } else {
      const float g = gamma[n];
#pragma unroll
      for (int r = 0; r < 8; ++r) {
        const int m = m0 + r + half * 8;
        const size_t idx = (size_t)m * N + n;
        outf[idx] = res[idx] + g * (acc[j][r] + bn);
      }
    }
  }
}

// scores tile: sc[c] (c = s-chunk of 16) = q[16xD] @ k^T, D=64, K in LDS (s,d)
__device__ __forceinline__ void score_tile(const _Float16* __restrict__ qrow,
                                           const _Float16* ldsK, v8f sc[4]) {
#pragma unroll
  for (int kk = 0; kk < HEAD_DIM; kk += 32) {
    v16h a = load_tile(qrow + kk, EMBED, 1);
#pragma unroll
    for (int c = 0; c < 4; ++c) {
      v16h bt = load_tile(ldsK + c * 16 * LDP + kk, LDP, 1);
      sc[c] = wmma16(a, bt, sc[c]);
    }
  }
}

// ---------------------------------------------------------------------------
// v-side attention, fused: per (b,h,128 t-rows) compute scores, row softmax,
// out_v = P @ Vl.  grid = (T/128, H, B), 256 threads (8 waves x 16 rows)
// ---------------------------------------------------------------------------
__global__ __launch_bounds__(256)
void attn_v_kernel(const _Float16* __restrict__ qh,
                   const _Float16* __restrict__ kh,
                   const _Float16* __restrict__ vlh,
                   _Float16* __restrict__ outv) {
  __shared__ _Float16 ldsK[64 * LDP];
  __shared__ _Float16 ldsV[64 * LDP];
  __shared__ _Float16 ldsP[8 * 16 * LDP];
  const int b = blockIdx.z, h = blockIdx.y;
  const int tid = threadIdx.x;
  const int wave = tid >> 5, lane = tid & 31, half = lane >> 4, l15 = lane & 15;

  for (int i = tid; i < 64 * 64; i += 256) {
    const int s = i >> 6, d = i & 63;
    const size_t g = (size_t)(b * SRC + s) * EMBED + h * HEAD_DIM + d;
    ldsK[s * LDP + d] = kh[g];
    ldsV[s * LDP + d] = vlh[g];
  }
  __syncthreads();

  const int t0 = blockIdx.x * 128 + wave * 16;
  const _Float16* qrow = qh + (size_t)(b * TGT + t0) * EMBED + h * HEAD_DIM;
  v8f z = {};
  v8f sc[4] = {z, z, z, z};
  score_tile(qrow, ldsK, sc);

  // row softmax (rows in regs, 64 cols across 16 lanes x 4 chunks)
  float inv[8];
#pragma unroll
  for (int r = 0; r < 8; ++r) {
    float mx = fmaxf(fmaxf(sc[0][r], sc[1][r]), fmaxf(sc[2][r], sc[3][r]));
    mx = redmax16(mx);
#pragma unroll
    for (int c = 0; c < 4; ++c) sc[c][r] = __expf(sc[c][r] - mx);
    float sm = sc[0][r] + sc[1][r] + sc[2][r] + sc[3][r];
    sm = redsum16(sm);
    inv[r] = 1.f / sm;
  }
  _Float16* pw = ldsP + wave * 16 * LDP;   // per-wave staging (C->A layout)
#pragma unroll
  for (int r = 0; r < 8; ++r) {
    const int m = r + half * 8;
#pragma unroll
    for (int c = 0; c < 4; ++c)
      pw[m * LDP + c * 16 + l15] = (_Float16)(sc[c][r] * inv[r]);
  }
  __syncthreads();

  v8f oc[4] = {z, z, z, z};
#pragma unroll
  for (int kk = 0; kk < SRC; kk += 32) {
    v16h a = load_tile(pw + kk, LDP, 1);
#pragma unroll
    for (int c = 0; c < 4; ++c) {
      v16h bv = load_tile(ldsV + kk * LDP + c * 16, 1, LDP);
      oc[c] = wmma16(a, bv, oc[c]);
    }
  }
#pragma unroll
  for (int r = 0; r < 8; ++r) {
    const int t = t0 + r + half * 8;
#pragma unroll
    for (int c = 0; c < 4; ++c)
      outv[(size_t)(b * TGT + t) * EMBED + h * HEAD_DIM + c * 16 + l15] =
          (_Float16)(oc[c][r]);
  }
}

// ---------------------------------------------------------------------------
// l-side attention, split-T partials: per (b,h,z) block handles T-range
// [z*T/TSPLIT, ...). Pass 1: column (s) max/expsum stats over the range.
// Pass 2: partial acc_z[s,d] = sum_t exp(sc - m_z) * vv[t,d] via WMMA.
// grid = (H, B, TSPLIT), 256 threads
// ---------------------------------------------------------------------------
__global__ __launch_bounds__(256)
void attn_l_partial_kernel(const _Float16* __restrict__ qh,
                           const _Float16* __restrict__ kh,
                           const _Float16* __restrict__ vvh,
                           float* __restrict__ pm, float* __restrict__ pl,
                           float* __restrict__ pacc) {
  __shared__ _Float16 ldsK[64 * LDP];
  __shared__ _Float16 ldsP[128 * LDP];
  __shared__ _Float16 ldsVV[128 * LDP];
  __shared__ float wm[8][64], wl[8][64];
  __shared__ float sm_[64];
  const int b = blockIdx.y, h = blockIdx.x, tz = blockIdx.z;
  const int tbeg = tz * (TGT / TSPLIT), tend = tbeg + (TGT / TSPLIT);
  const int part = (b * HEADS + h) * TSPLIT + tz;
  const int tid = threadIdx.x;
  const int wave = tid >> 5, lane = tid & 31, half = lane >> 4, l15 = lane & 15;

  for (int i = tid; i < 64 * 64; i += 256) {
    const int s = i >> 6, d = i & 63;
    ldsK[s * LDP + d] = kh[(size_t)(b * SRC + s) * EMBED + h * HEAD_DIM + d];
  }
  __syncthreads();

  // ---- pass 1: per-column (s) max & expsum over this T-range ----
  float mloc[4] = {-1e30f, -1e30f, -1e30f, -1e30f};
  float lloc[4] = {0.f, 0.f, 0.f, 0.f};
  v8f z = {};
  for (int t0 = tbeg + wave * 16; t0 < tend; t0 += 128) {
    const _Float16* qrow = qh + (size_t)(b * TGT + t0) * EMBED + h * HEAD_DIM;
    v8f sc[4] = {z, z, z, z};
    score_tile(qrow, ldsK, sc);
#pragma unroll
    for (int c = 0; c < 4; ++c) {
      float tm = sc[c][0];
#pragma unroll
      for (int r = 1; r < 8; ++r) tm = fmaxf(tm, sc[c][r]);
      const float nm = fmaxf(mloc[c], tm);
      float add = 0.f;
#pragma unroll
      for (int r = 0; r < 8; ++r) add += __expf(sc[c][r] - nm);
      lloc[c] = lloc[c] * __expf(mloc[c] - nm) + add;
      mloc[c] = nm;
    }
  }
#pragma unroll
  for (int c = 0; c < 4; ++c) {   // merge halves (rows split across halves)
    const float om = lane_xor<16>(mloc[c]);
    const float ol = lane_xor<16>(lloc[c]);
    const float nm = fmaxf(mloc[c], om);
    const float nl = lloc[c] * __expf(mloc[c] - nm) + ol * __expf(om - nm);
    if (half == 0) { wm[wave][c * 16 + l15] = nm; wl[wave][c * 16 + l15] = nl; }
  }
  __syncthreads();
  if (tid < 64) {                  // merge across the 8 waves
    float m = wm[0][tid], l = wl[0][tid];
#pragma unroll
    for (int w = 1; w < 8; ++w) {
      const float om = wm[w][tid], ol = wl[w][tid];
      const float nm = fmaxf(m, om);
      l = l * __expf(m - nm) + ol * __expf(om - nm);
      m = nm;
    }
    sm_[tid] = m;
    pm[part * 64 + tid] = m;
    pl[part * 64 + tid] = l;
  }
  __syncthreads();
  float cm[4];
#pragma unroll
  for (int c = 0; c < 4; ++c) cm[c] = sm_[c * 16 + l15];

  // ---- pass 2: acc_z[s,d] = sum_t exp(sc - m_z) * vv[t,d] ----
  const int s0 = (wave & 3) * 16, d0 = (wave >> 2) * 32;
  v8f accl[2] = {z, z};
  for (int t0 = tbeg; t0 < tend; t0 += 128) {
    // stage vv tile (128 x 64) cooperatively
    for (int i = tid; i < 128 * 16; i += 256) {
      const int rr = i >> 4, cu = i & 15;
      const uint2* src = (const uint2*)(vvh + (size_t)(b * TGT + t0 + rr) * EMBED +
                                        h * HEAD_DIM);
      ((uint2*)(ldsVV + rr * LDP))[cu] = src[cu];
    }
    // recompute scores for this wave's 16 rows, exponentiate, stage P
    const int tw = t0 + wave * 16;
    const _Float16* qrow = qh + (size_t)(b * TGT + tw) * EMBED + h * HEAD_DIM;
    v8f sc[4] = {z, z, z, z};
    score_tile(qrow, ldsK, sc);
#pragma unroll
    for (int r = 0; r < 8; ++r) {
      const int m = wave * 16 + r + half * 8;
#pragma unroll
      for (int c = 0; c < 4; ++c)
        ldsP[m * LDP + c * 16 + l15] = (_Float16)__expf(sc[c][r] - cm[c]);
    }
    __syncthreads();
    // accumulate: A = P^T (s x t), B = vv (t x d)
#pragma unroll
    for (int kk = 0; kk < 128; kk += 32) {
      v16h a = load_tile(ldsP + kk * LDP + s0, 1, LDP);
#pragma unroll
      for (int cc = 0; cc < 2; ++cc) {
        v16h bv = load_tile(ldsVV + kk * LDP + d0 + cc * 16, 1, LDP);
        accl[cc] = wmma16(a, bv, accl[cc]);
      }
    }
    __syncthreads();
  }
#pragma unroll
  for (int r = 0; r < 8; ++r) {
    const int s = s0 + r + half * 8;
#pragma unroll
    for (int cc = 0; cc < 2; ++cc)
      pacc[(size_t)part * 4096 + s * 64 + d0 + cc * 16 + l15] = accl[cc][r];
  }
}

// Combine TSPLIT partials: out_l[s,d] = sum_z acc_z[s,d]*e^(m_z-M) / L
// grid = B*H blocks, 256 threads
__global__ __launch_bounds__(256)
void attn_l_combine_kernel(const float* __restrict__ pm,
                           const float* __restrict__ pl,
                           const float* __restrict__ pacc,
                           _Float16* __restrict__ outl) {
  const int bh = blockIdx.x;
  const int b = bh / HEADS, h = bh - b * HEADS;
  __shared__ float fz[TSPLIT][64];
  __shared__ float invL[64];
  const int tid = threadIdx.x;
  if (tid < 64) {
    float M = -1e30f;
#pragma unroll
    for (int zi = 0; zi < TSPLIT; ++zi)
      M = fmaxf(M, pm[(bh * TSPLIT + zi) * 64 + tid]);
    float L = 0.f;
#pragma unroll
    for (int zi = 0; zi < TSPLIT; ++zi) {
      const float f = __expf(pm[(bh * TSPLIT + zi) * 64 + tid] - M);
      fz[zi][tid] = f;
      L += pl[(bh * TSPLIT + zi) * 64 + tid] * f;
    }
    invL[tid] = 1.f / L;
  }
  __syncthreads();
  for (int i = tid; i < 4096; i += 256) {
    const int s = i >> 6, d = i & 63;
    float acc = 0.f;
#pragma unroll
    for (int zi = 0; zi < TSPLIT; ++zi)
      acc += pacc[(size_t)(bh * TSPLIT + zi) * 4096 + i] * fz[zi][s];
    outl[(size_t)(b * SRC + s) * EMBED + h * HEAD_DIM + d] =
        (_Float16)(acc * invL[s]);
  }
}

// ---------------------------------------------------------------------------
extern "C" void kernel_launch(void* const* d_in, const int* in_sizes, int n_in,
                              void* d_out, int out_size, void* d_ws,
                              size_t ws_size, hipStream_t stream) {
  (void)in_sizes; (void)n_in; (void)out_size; (void)ws_size;
  const float* v      = (const float*)d_in[0];
  const float* l      = (const float*)d_in[1];
  const float* v_pos  = (const float*)d_in[2];
  const float* ln_v_w = (const float*)d_in[3];
  const float* ln_v_b = (const float*)d_in[4];
  const float* ln_l_w = (const float*)d_in[5];
  const float* ln_l_b = (const float*)d_in[6];
  const float* w_q  = (const float*)d_in[7];  const float* b_q  = (const float*)d_in[8];
  const float* w_k  = (const float*)d_in[9];  const float* b_k  = (const float*)d_in[10];
  const float* w_vv = (const float*)d_in[11]; const float* b_vv = (const float*)d_in[12];
  const float* w_vl = (const float*)d_in[13]; const float* b_vl = (const float*)d_in[14];
  const float* w_ov = (const float*)d_in[15]; const float* b_ov = (const float*)d_in[16];
  const float* w_ol = (const float*)d_in[17]; const float* b_ol = (const float*)d_in[18];
  const float* gamma_v = (const float*)d_in[19];
  const float* gamma_l = (const float*)d_in[20];

  const int MT = B_ * TGT;   // 65536
  const int ML = B_ * SRC;   // 512

  _Float16* ws = (_Float16*)d_ws;
  size_t off = 0;
  _Float16* vn_h  = ws + off; off += (size_t)MT * V_DIM;
  _Float16* qin_h = ws + off; off += (size_t)MT * V_DIM;
  _Float16* ln_h  = ws + off; off += (size_t)ML * L_DIM;
  _Float16* q_h   = ws + off; off += (size_t)MT * EMBED;
  _Float16* vv_h  = ws + off; off += (size_t)MT * EMBED;
  _Float16* k_h   = ws + off; off += (size_t)ML * EMBED;
  _Float16* vl_h  = ws + off; off += (size_t)ML * EMBED;
  _Float16* ov_h  = ws + off; off += (size_t)MT * EMBED;
  _Float16* ol_h  = ws + off; off += (size_t)ML * EMBED;
  _Float16* wt_q  = ws + off; off += (size_t)V_DIM * EMBED;
  _Float16* wt_k  = ws + off; off += (size_t)L_DIM * EMBED;
  _Float16* wt_vv = ws + off; off += (size_t)V_DIM * EMBED;
  _Float16* wt_vl = ws + off; off += (size_t)L_DIM * EMBED;
  _Float16* wt_ov = ws + off; off += (size_t)EMBED * V_DIM;
  _Float16* wt_ol = ws + off; off += (size_t)EMBED * L_DIM;
  // f32 scratch for split-T l-side attention (even half-offsets => 4B aligned)
  float* pm   = (float*)(ws + off); off += (size_t)B_ * HEADS * TSPLIT * 64 * 2;
  float* pl   = (float*)(ws + off); off += (size_t)B_ * HEADS * TSPLIT * 64 * 2;
  float* pacc = (float*)(ws + off); off += (size_t)B_ * HEADS * TSPLIT * 4096 * 2;

  float* out_v = (float*)d_out;
  float* out_l = (float*)d_out + (size_t)MT * V_DIM;

  // 1) weights -> f16, transposed to [N,K]
  auto cw = [&](const float* w, _Float16* wt, int K, int N) {
    convert_wT_kernel<<<(K * N + 255) / 256, 256, 0, stream>>>(w, wt, K, N);
  };
  cw(w_q,  wt_q,  V_DIM, EMBED);
  cw(w_k,  wt_k,  L_DIM, EMBED);
  cw(w_vv, wt_vv, V_DIM, EMBED);
  cw(w_vl, wt_vl, L_DIM, EMBED);
  cw(w_ov, wt_ov, EMBED, V_DIM);
  cw(w_ol, wt_ol, EMBED, L_DIM);

  // 2) layernorms (v also fuses +v_pos query input)
  layernorm_kernel<V_DIM><<<MT / 8, 256, 0, stream>>>(v, v_pos, ln_v_w, ln_v_b,
                                                      vn_h, qin_h, MT);
  layernorm_kernel<L_DIM><<<ML / 8, 256, 0, stream>>>(l, nullptr, ln_l_w, ln_l_b,
                                                      ln_h, nullptr, ML);

  // 3) projections (WMMA GEMMs), q carries *SCALE after bias
  gemm_wmma_kernel<0><<<dim3(MT / 128, EMBED / 64), 256, 0, stream>>>(
      qin_h, wt_q, b_q, MT, EMBED, V_DIM, q_h, nullptr, nullptr, nullptr, QSCALE);
  gemm_wmma_kernel<0><<<dim3(ML / 128, EMBED / 64), 256, 0, stream>>>(
      ln_h, wt_k, b_k, ML, EMBED, L_DIM, k_h, nullptr, nullptr, nullptr, 1.f);
  gemm_wmma_kernel<0><<<dim3(MT / 128, EMBED / 64), 256, 0, stream>>>(
      vn_h, wt_vv, b_vv, MT, EMBED, V_DIM, vv_h, nullptr, nullptr, nullptr, 1.f);
  gemm_wmma_kernel<0><<<dim3(ML / 128, EMBED / 64), 256, 0, stream>>>(
      ln_h, wt_vl, b_vl, ML, EMBED, L_DIM, vl_h, nullptr, nullptr, nullptr, 1.f);

  // 4) attention (both directions)
  attn_v_kernel<<<dim3(TGT / 128, HEADS, B_), 256, 0, stream>>>(q_h, k_h, vl_h, ov_h);
  attn_l_partial_kernel<<<dim3(HEADS, B_, TSPLIT), 256, 0, stream>>>(
      q_h, k_h, vv_h, pm, pl, pacc);
  attn_l_combine_kernel<<<B_ * HEADS, 256, 0, stream>>>(pm, pl, pacc, ol_h);

  // 5) output projections + gated residual -> d_out (f32)
  gemm_wmma_kernel<1><<<dim3(MT / 128, V_DIM / 64), 256, 0, stream>>>(
      ov_h, wt_ov, b_ov, MT, V_DIM, EMBED, nullptr, out_v, v, gamma_v, 1.f);
  gemm_wmma_kernel<1><<<dim3(ML / 128, L_DIM / 64), 256, 0, stream>>>(
      ol_h, wt_ol, b_ol, ML, L_DIM, EMBED, nullptr, out_l, l, gamma_l, 1.f);
}
